// GINAttenSyn_8735963480682
// MI455X (gfx1250) — matrix operand order repaired and checked by reading, not compiled, not run on hardware
//
#include <hip/hip_runtime.h>
#include <hip/hip_bf16.h>
#include <math.h>

// ---------------------------------------------------------------------------
// GINAttenSyn forward for MI455X (gfx1250, wave32).
// Dense GEMMs: v_wmma_f32_16x16x32_f16, 16x32 C-tile per wave (2 wmma/K-step),
// unconditional float4 A-loads + pre-swizzled f16 B-fragments (v16h loads).
// Graph aggregation via global_atomic_add_f32; LSTM recurrence is a
// single-block LDS scan (latency-bound, not throughput-bound).
// ---------------------------------------------------------------------------

#define NNODES 25600
#define CIN    78
#define CINP   96      // CIN padded to K-tile multiple (32)
#define HHID   64
#define NEDGE  204800
#define NB     256
#define NPG    100
#define NCELL  954
#define NCELLP 960     // NCELL padded to 32
#define EPSV   1e-5f

typedef __attribute__((ext_vector_type(16))) _Float16 v16h;
typedef __attribute__((ext_vector_type(8)))  float    v8f;

static inline int ceil_div_i(int a, int b) { return (a + b - 1) / b; }

// ------------------------------ utility kernels ----------------------------

__global__ void k_zero(float* __restrict__ p, int n) {
    int i = blockIdx.x * blockDim.x + threadIdx.x;
    if (i < n) p[i] = 0.0f;
}

__global__ void k_bias_comb(const float* __restrict__ a, const float* __restrict__ b,
                            float* __restrict__ c) {
    int i = threadIdx.x;             // 256 threads
    c[i] = a[i] + b[i];
}

__global__ void k_padbias(const float* __restrict__ s, float* __restrict__ d,
                          int n, int npad) {
    int i = blockIdx.x * blockDim.x + threadIdx.x;
    if (i < npad) d[i] = (i < n) ? s[i] : 0.0f;
}

// row-wise L2 normalize cell (B x NCELL) -> out stride NCELLP (pads pre-zeroed)
__global__ void k_rownorm(const float* __restrict__ cell, float* __restrict__ out) {
    int b = blockIdx.x, tid = threadIdx.x;
    __shared__ float red[256];
    __shared__ float nf;
    float s = 0.0f;
    for (int i = tid; i < NCELL; i += 256) {
        float v = cell[(size_t)b * NCELL + i];
        s += v * v;
    }
    red[tid] = s; __syncthreads();
    for (int st = 128; st > 0; st >>= 1) {
        if (tid < st) red[tid] += red[tid + st];
        __syncthreads();
    }
    if (tid == 0) nf = 1.0f / fmaxf(sqrtf(red[0]), 1e-12f);
    __syncthreads();
    for (int i = tid; i < NCELL; i += 256)
        out[(size_t)b * NCELLP + i] = cell[(size_t)b * NCELL + i] * nf;
}

// hin[t, c] (stride CINP) = x[t, c] (stride CIN) + cell78[t/NPG, c] (stride CINP)
__global__ void k_cellexp(const float* __restrict__ x, const float* __restrict__ c78,
                          float* __restrict__ hin, int n) {
    int i = blockIdx.x * blockDim.x + threadIdx.x;
    if (i >= n) return;
    int t = i / CIN, c = i - t * CIN;
    hin[(size_t)t * CINP + c] = x[i] + c78[(size_t)(t / NPG) * CINP + c];
}

// segment_sum: agg[dst[e], c] += x[src[e], c]  (row stride ld, K channels)
__global__ void k_scatter(const float* __restrict__ x, const int* __restrict__ ei,
                          float* __restrict__ agg, int K, int ld) {
    long long i = (long long)blockIdx.x * blockDim.x + threadIdx.x;
    long long tot = (long long)NEDGE * K;
    if (i >= tot) return;
    int e = (int)(i / K);
    int c = (int)(i - (long long)e * K);
    int s = ei[e];
    int d = ei[NEDGE + e];
    atomicAdd(&agg[(size_t)d * ld + c], x[(size_t)s * ld + c]);
}

// c[r, cc] = a[r, cc] + b[r, cc] over rows x K, common stride ld
__global__ void k_add_rows(const float* __restrict__ a, const float* __restrict__ b,
                           float* __restrict__ c, int n, int K, int ld) {
    int i = blockIdx.x * blockDim.x + threadIdx.x;
    if (i >= n) return;
    int r = i / K, cc = i - r * K;
    size_t off = (size_t)r * ld + cc;
    c[off] = a[off] + b[off];
}

// ---------------------- weight -> B-fragment swizzle -----------------------
// dst element idx = ((kt*ntN + nt)*32 + lane)*16 + i  holds  W[kt*32 + hi*8 + (i&7) + (i>>3)*16,
// nt*16 + (lane&15)] as f16 (zero-padded past K/Nc). One v16h per (tile, lane).
__global__ void k_wlayout(const float* __restrict__ W, _Float16* __restrict__ dst,
                          int K, int Nc, int ntN, int total) {
    int idx = blockIdx.x * blockDim.x + threadIdx.x;
    if (idx >= total) return;
    int i    = idx & 15;
    int lane = (idx >> 4) & 31;
    int t    = idx >> 9;
    int kt   = t / ntN;
    int nt   = t - kt * ntN;
    int hi   = lane >> 4;
    int k    = kt * 32 + hi * 8 + (i & 7) + ((i >> 3) << 4);
    int col  = nt * 16 + (lane & 15);
    dst[idx] = (k < K && col < Nc) ? (_Float16)W[(size_t)k * Nc + col] : (_Float16)0.0f;
}

// ------------------------------ WMMA GEMM ----------------------------------
// C[M, Nc32*32] = relu?( A[M, KT*32](f32, zero-padded) @ Wfrag + biasPad )
// One wave -> 16x32 C tile: one A fragment, two B fragments, two wmma / K-step.
__global__ void k_gemm_wmma(const float* __restrict__ A, int lda,
                            const _Float16* __restrict__ Wraw,
                            const float* __restrict__ bias,
                            float* __restrict__ C,
                            int M, int Nc32, int KT, int relu) {
    const int tiles = (M >> 4) * Nc32;
    int tile = blockIdx.x * 4 + (threadIdx.x >> 5);   // 4 waves / block
    if (tile >= tiles) return;                        // wave-uniform exit
    const int tM = tile / Nc32, tP = tile - tM * Nc32;
    const int lane = threadIdx.x & 31;
    const int l15  = lane & 15;
    const int hi   = lane >> 4;
    const int row  = tM * 16 + l15;
    const int ntN  = Nc32 * 2;
    const int ldc  = Nc32 * 32;

    const float4* __restrict__ A4 = (const float4*)(A + (size_t)row * lda);
    const v16h*   __restrict__ Wf = (const v16h*)Wraw;

    v8f acc0 = {}, acc1 = {};
    for (int kt = 0; kt < KT; ++kt) {
        const int kb4 = kt * 8 + hi * 2;              // float4 index of kbase+hi*8
        float4 p0 = A4[kb4];
        float4 p1 = A4[kb4 + 1];
        float4 p2 = A4[kb4 + 4];                      // kbase + 16
        float4 p3 = A4[kb4 + 5];
        v16h a;
        a[0]  = (_Float16)p0.x; a[1]  = (_Float16)p0.y;
        a[2]  = (_Float16)p0.z; a[3]  = (_Float16)p0.w;
        a[4]  = (_Float16)p1.x; a[5]  = (_Float16)p1.y;
        a[6]  = (_Float16)p1.z; a[7]  = (_Float16)p1.w;
        a[8]  = (_Float16)p2.x; a[9]  = (_Float16)p2.y;
        a[10] = (_Float16)p2.z; a[11] = (_Float16)p2.w;
        a[12] = (_Float16)p3.x; a[13] = (_Float16)p3.y;
        a[14] = (_Float16)p3.z; a[15] = (_Float16)p3.w;
        v16h b0 = Wf[((size_t)kt * ntN + 2 * tP)     * 32 + lane];
        v16h b1 = Wf[((size_t)kt * ntN + 2 * tP + 1) * 32 + lane];
        if (kt + 1 < KT)                              // global_prefetch_b8 path
            __builtin_prefetch(A4 + kb4 + 8, 0, 1);
        acc0 = __builtin_amdgcn_wmma_f32_16x16x32_f16(false, a, false, b0,
                                                      (short)0, acc0, false, false);
        acc1 = __builtin_amdgcn_wmma_f32_16x16x32_f16(false, a, false, b1,
                                                      (short)0, acc1, false, false);
    }
    const int col0 = tP * 32 + l15;
    const int col1 = col0 + 16;
    const float b0 = bias[col0], b1 = bias[col1];
    #pragma unroll
    for (int r = 0; r < 8; ++r) {
        int rr = tM * 16 + r + hi * 8;                // C/D layout: VGPR r -> M=r (+8 upper)
        float v0 = acc0[r] + b0;
        float v1 = acc1[r] + b1;
        if (relu) { v0 = fmaxf(v0, 0.0f); v1 = fmaxf(v1, 0.0f); }
        C[(size_t)rr * ldc + col0] = v0;
        C[(size_t)rr * ldc + col1] = v1;
    }
}

// ------------------------------ BatchNorm ----------------------------------

__global__ void k_bnstats(const float* __restrict__ h, float* __restrict__ stats) {
    int ch = blockIdx.x, tid = threadIdx.x;           // 64 blocks x 256 threads
    __shared__ float s1[256], s2[256];
    float a = 0.0f, b = 0.0f;
    for (int t = tid; t < NNODES; t += 256) {
        float v = h[(size_t)t * HHID + ch];
        a += v; b += v * v;
    }
    s1[tid] = a; s2[tid] = b; __syncthreads();
    for (int st = 128; st > 0; st >>= 1) {
        if (tid < st) { s1[tid] += s1[tid + st]; s2[tid] += s2[tid + st]; }
        __syncthreads();
    }
    if (tid == 0) {
        float mu  = s1[0] / (float)NNODES;
        float var = s2[0] / (float)NNODES - mu * mu;
        stats[2 * ch]     = mu;
        stats[2 * ch + 1] = rsqrtf(var + EPSV);
    }
}

__global__ void k_bnapply(const float* __restrict__ hpre, const float* __restrict__ stats,
                          const float* __restrict__ g, const float* __restrict__ beta,
                          const float* __restrict__ res, float* __restrict__ out, int n) {
    int i = blockIdx.x * blockDim.x + threadIdx.x;
    if (i >= n) return;
    int ch = i & (HHID - 1);
    float v = (hpre[i] - stats[2 * ch]) * stats[2 * ch + 1] * g[ch] + beta[ch];
    if (res) v += res[i];
    out[i] = fmaxf(v, 0.0f);
}

// ------------------------------ LSTM scan ----------------------------------
__global__ void k_lstm(const float* __restrict__ gates,
                       const float* __restrict__ whh,   // (64,256) row-major
                       float* __restrict__ state,       // h[64], c[64] persists
                       float* __restrict__ rnn) {       // (N,64) out
    __shared__ float hsh[HHID], csh[HHID], gsh[256];
    int tid = threadIdx.x;                             // 256 threads
    if (tid < HHID) { hsh[tid] = state[tid]; csh[tid] = state[HHID + tid]; }
    __syncthreads();
    for (int t = 0; t < NNODES; ++t) {
        const float* g = gates + (size_t)t * 256;
        float acc = g[tid];
        #pragma unroll 8
        for (int d = 0; d < HHID; ++d) acc += hsh[d] * whh[d * 256 + tid];
        gsh[tid] = acc;
        __syncthreads();
        if (tid < HHID) {
            float ig = 1.0f / (1.0f + __expf(-gsh[tid]));
            float fg = 1.0f / (1.0f + __expf(-gsh[64 + tid]));
            float gg = tanhf(gsh[128 + tid]);
            float og = 1.0f / (1.0f + __expf(-gsh[192 + tid]));
            float c  = fg * csh[tid] + ig * gg;
            float h  = og * tanhf(c);
            csh[tid] = c; hsh[tid] = h;
            rnn[(size_t)t * HHID + tid] = h;
        }
        __syncthreads();
    }
    if (tid < HHID) { state[tid] = hsh[tid]; state[HHID + tid] = csh[tid]; }
}

// ------------------------------ Attention pool -----------------------------
__global__ void k_attn(const float* __restrict__ ksf, const float* __restrict__ qot,
                       const float* __restrict__ vsf, const float* __restrict__ ssf,
                       const unsigned char* __restrict__ mask,
                       const float* __restrict__ lng, const float* __restrict__ lnb,
                       float* __restrict__ outb, int colOff) {
    int b = blockIdx.x, tid = threadIdx.x;             // 128 threads
    __shared__ float qsh[NPG * HHID];
    __shared__ float sc[NPG];
    __shared__ float red[2];
    __shared__ float vec[HHID];
    for (int i = tid; i < NPG * HHID; i += blockDim.x)
        qsh[i] = qot[(size_t)b * NPG * HHID + i];
    __syncthreads();
    if (tid < NPG) {
        const float* kr = ksf + ((size_t)b * NPG + tid) * HHID;
        float s = 0.0f;
        for (int m = 0; m < NPG; ++m) {
            float acc = 0.0f;
            for (int d = 0; d < HHID; ++d) acc += kr[d] * qsh[m * HHID + d];
            s += tanhf(acc);
        }
        if (mask[b * NPG + tid]) s = -__builtin_inff();
        sc[tid] = s;
    }
    __syncthreads();
    if (tid == 0) {
        float mx = -__builtin_inff();
        for (int n = 0; n < NPG; ++n) mx = fmaxf(mx, sc[n]);
        float sum = 0.0f;
        for (int n = 0; n < NPG; ++n) { sc[n] = __expf(sc[n] - mx); sum += sc[n]; }
        red[0] = sum;
    }
    __syncthreads();
    float inv = 1.0f / red[0];
    if (tid < HHID) {
        float acc = 0.0f, mn = 0.0f;
        for (int n = 0; n < NPG; ++n) {
            size_t basei = ((size_t)b * NPG + n) * HHID + tid;
            acc += sc[n] * inv * vsf[basei];
            mn  += ssf[basei];
        }
        vec[tid] = mn / (float)NPG + acc;
    }
    __syncthreads();
    if (tid == 0) {
        float mu = 0.0f;
        for (int d = 0; d < HHID; ++d) mu += vec[d];
        mu /= (float)HHID;
        float var = 0.0f;
        for (int d = 0; d < HHID; ++d) { float t = vec[d] - mu; var += t * t; }
        var /= (float)HHID;
        red[0] = mu; red[1] = rsqrtf(var + EPSV);
    }
    __syncthreads();
    if (tid < HHID)
        outb[(size_t)b * 512 + colOff + tid] =
            (vec[tid] - red[0]) * red[1] * lng[tid] + lnb[tid];
}

// concat[b, 256+c] = cell256[b, c]
__global__ void k_copy256(const float* __restrict__ c256, float* __restrict__ cat) {
    int i = blockIdx.x * blockDim.x + threadIdx.x;
    if (i >= NB * 256) return;
    int b = i >> 8, c = i & 255;
    cat[(size_t)b * 512 + 256 + c] = c256[i];
}

__global__ void k_fin2(const float* __restrict__ mid, const float* __restrict__ w2,
                       const float* __restrict__ b2, float* __restrict__ out) {
    int i = blockIdx.x * blockDim.x + threadIdx.x;
    if (i >= NB * 2) return;
    int b = i >> 1, j = i & 1;
    float acc = b2[j];
    for (int d = 0; d < HHID; ++d) acc += mid[b * HHID + d] * w2[d * 2 + j];
    out[i] = acc;
}

// ------------------------------ host driver --------------------------------

extern "C" void kernel_launch(void* const* d_in, const int* in_sizes, int n_in,
                              void* d_out, int out_size, void* d_ws, size_t ws_size,
                              hipStream_t stream) {
    (void)in_sizes; (void)n_in; (void)out_size; (void)ws_size;
    const float* x1   = (const float*)d_in[0];
    const float* x2   = (const float*)d_in[1];
    const int*   ei1  = (const int*)d_in[2];
    const int*   ei2  = (const int*)d_in[3];
    const unsigned char* m1 = (const unsigned char*)d_in[6];
    const unsigned char* m2 = (const unsigned char*)d_in[7];
    const float* cell = (const float*)d_in[8];
    const float* g0w1 = (const float*)d_in[9],  *g0b1 = (const float*)d_in[10];
    const float* g0w2 = (const float*)d_in[11], *g0b2 = (const float*)d_in[12];
    const float* g1w1 = (const float*)d_in[13], *g1b1 = (const float*)d_in[14];
    const float* g1w2 = (const float*)d_in[15], *g1b2 = (const float*)d_in[16];
    const float* bn0g = (const float*)d_in[17], *bn0b = (const float*)d_in[18];
    const float* bn1g = (const float*)d_in[19], *bn1b = (const float*)d_in[20];
    const float* wih  = (const float*)d_in[21], *whh  = (const float*)d_in[22];
    const float* bih  = (const float*)d_in[23], *bhh  = (const float*)d_in[24];
    const float* r1w1 = (const float*)d_in[25], *r1b1 = (const float*)d_in[26];
    const float* r1w2 = (const float*)d_in[27], *r1b2 = (const float*)d_in[28];
    const float* r2w1 = (const float*)d_in[29], *r2b1 = (const float*)d_in[30];
    const float* r2w2 = (const float*)d_in[31], *r2b2 = (const float*)d_in[32];
    const float* p1wq = (const float*)d_in[33], *p1bq = (const float*)d_in[34];
    const float* p1wk = (const float*)d_in[35], *p1bk = (const float*)d_in[36];
    const float* p1wv = (const float*)d_in[37], *p1bv = (const float*)d_in[38];
    const float* p1g  = (const float*)d_in[39], *p1b  = (const float*)d_in[40];
    const float* p2wq = (const float*)d_in[41], *p2bq = (const float*)d_in[42];
    const float* p2wk = (const float*)d_in[43], *p2bk = (const float*)d_in[44];
    const float* p2wv = (const float*)d_in[45], *p2bv = (const float*)d_in[46];
    const float* p2g  = (const float*)d_in[47], *p2b  = (const float*)d_in[48];
    const float* fw1  = (const float*)d_in[49], *fb1  = (const float*)d_in[50];
    const float* fw2  = (const float*)d_in[51], *fb2  = (const float*)d_in[52];
    float* out = (float*)d_out;

    // bump allocator on d_ws (256B aligned blocks)
    char* base = (char*)d_ws; size_t off = 0;
    auto allocF = [&](size_t n) -> float* {
        float* p = (float*)(base + off);
        off = (off + n * sizeof(float) + 255) & ~(size_t)255;
        return p;
    };
    auto allocH = [&](size_t n) -> _Float16* {
        _Float16* p = (_Float16*)(base + off);
        off = (off + n * sizeof(_Float16) + 255) & ~(size_t)255;
        return p;
    };

    float* celln  = allocF((size_t)NB * NCELLP);       // stride 960, pads zero
    float* t512   = allocF((size_t)NB * 512);
    float* c256   = allocF((size_t)NB * 256);
    float* c78    = allocF((size_t)NB * CINP);         // stride 96 (GEMM-padded out)
    float* hin_l  = allocF((size_t)NNODES * CINP);     // stride 96, pads zero
    float* hin_r  = allocF((size_t)NNODES * CINP);
    float* agg    = allocF((size_t)NNODES * CINP);
    float* zbuf   = allocF((size_t)NNODES * CINP);
    float* mid    = allocF((size_t)NNODES * HHID);
    float* hpre   = allocF((size_t)NNODES * HHID);
    float* h_l    = allocF((size_t)NNODES * HHID);
    float* h_r    = allocF((size_t)NNODES * HHID);
    float* gates  = allocF((size_t)NNODES * 256);
    float* rnn_l  = allocF((size_t)NNODES * HHID);
    float* rnn_r  = allocF((size_t)NNODES * HHID);
    float* q1 = allocF((size_t)NNODES * HHID); float* k1 = allocF((size_t)NNODES * HHID);
    float* v1 = allocF((size_t)NNODES * HHID); float* q2 = allocF((size_t)NNODES * HHID);
    float* k2 = allocF((size_t)NNODES * HHID); float* v2 = allocF((size_t)NNODES * HHID);
    float* stats  = allocF(128);
    float* st_l   = allocF(128);
    float* st_r   = allocF(128);
    float* catb   = allocF((size_t)NB * 512);
    float* finmid = allocF((size_t)NB * HHID);
    float* biasc  = allocF(256);
    float* r2b2p  = allocF(CINP);                      // padded bias for Nc=78

    auto EL = [&](long long n) { return dim3((unsigned)((n + 255) / 256)); };
    auto zero = [&](float* p, long long n) { k_zero<<<EL(n), 256, 0, stream>>>(p, (int)n); };

    // ---- weight fragment swizzle (B-layout, f16, zero-padded) ----
    struct WF { _Float16* p; int KT; int Nc32; };
    auto makeW = [&](const float* src, int K, int Nc) -> WF {
        int KT = ceil_div_i(K, 32), Nc32 = ceil_div_i(Nc, 32), ntN = Nc32 * 2;
        size_t n = (size_t)KT * ntN * 512;             // 32 lanes * 16 halves / 16-tile
        _Float16* p = allocH(n);
        k_wlayout<<<EL((long long)n), 256, 0, stream>>>(src, p, K, Nc, ntN, (int)n);
        return WF{p, KT, Nc32};
    };
    WF f_r1w1 = makeW(r1w1, NCELL, 512);
    WF f_r1w2 = makeW(r1w2, 512, 256);
    WF f_r2w1 = makeW(r2w1, NCELL, 512);
    WF f_r2w2 = makeW(r2w2, 512, CIN);
    WF f_g0w1 = makeW(g0w1, CIN, HHID);
    WF f_g0w2 = makeW(g0w2, HHID, HHID);
    WF f_g1w1 = makeW(g1w1, HHID, HHID);
    WF f_g1w2 = makeW(g1w2, HHID, HHID);
    WF f_wih  = makeW(wih, HHID, 256);
    WF f_p1q = makeW(p1wq, HHID, HHID), f_p1k = makeW(p1wk, HHID, HHID), f_p1v = makeW(p1wv, HHID, HHID);
    WF f_p2q = makeW(p2wq, HHID, HHID), f_p2k = makeW(p2wk, HHID, HHID), f_p2v = makeW(p2wv, HHID, HHID);
    WF f_fw1  = makeW(fw1, 512, HHID);

    auto gemm = [&](const float* A, int lda, const WF& w, const float* bias,
                    float* C, int M, int relu) {
        int tiles = (M / 16) * w.Nc32;
        k_gemm_wmma<<<ceil_div_i(tiles, 4), 128, 0, stream>>>(
            A, lda, w.p, bias, C, M, w.Nc32, w.KT, relu);
    };

    k_bias_comb<<<1, 256, 0, stream>>>(bih, bhh, biasc);
    k_padbias<<<1, CINP, 0, stream>>>(r2b2, r2b2p, CIN, CINP);

    // ---- cell reducers ----
    zero(celln, (long long)NB * NCELLP);
    k_rownorm<<<NB, 256, 0, stream>>>(cell, celln);
    gemm(celln, NCELLP, f_r1w1, r1b1, t512, NB, 1);
    gemm(t512, 512, f_r1w2, r1b2, c256, NB, 1);
    gemm(celln, NCELLP, f_r2w1, r2b1, t512, NB, 1);
    gemm(t512, 512, f_r2w2, r2b2p, c78, NB, 1);        // C stride 96 (padded)
    zero(hin_l, (long long)NNODES * CINP);
    zero(hin_r, (long long)NNODES * CINP);
    k_cellexp<<<EL((long long)NNODES * CIN), 256, 0, stream>>>(x1, c78, hin_l, NNODES * CIN);
    k_cellexp<<<EL((long long)NNODES * CIN), 256, 0, stream>>>(x2, c78, hin_r, NNODES * CIN);

    // ---- GIN layer (+BN+ReLU(+res)); xin row stride = ld (= KT*32, pads zero) ----
    auto run_gin = [&](const float* xin, int K, int ld, const int* ei,
                       const WF& w1f, const float* b1, const WF& w2f,
                       const float* b2, const float* bg, const float* bb,
                       const float* res, float* hout) {
        zero(agg, (long long)NNODES * ld);
        if (K != ld) zero(zbuf, (long long)NNODES * ld);   // refresh pad columns
        k_scatter<<<EL((long long)NEDGE * K), 256, 0, stream>>>(xin, ei, agg, K, ld);
        k_add_rows<<<EL((long long)NNODES * K), 256, 0, stream>>>(
            xin, agg, zbuf, NNODES * K, K, ld);
        gemm(zbuf, ld, w1f, b1, mid, NNODES, 1);
        gemm(mid, HHID, w2f, b2, hpre, NNODES, 0);
        k_bnstats<<<HHID, 256, 0, stream>>>(hpre, stats);
        k_bnapply<<<EL((long long)NNODES * HHID), 256, 0, stream>>>(
            hpre, stats, bg, bb, res, hout, NNODES * HHID);
    };

    // ---- side L: GIN0 -> LSTM0 -> GIN1 -> LSTM1 (state carried) ----
    run_gin(hin_l, CIN, CINP, ei1, f_g0w1, g0b1, f_g0w2, g0b2, bn0g, bn0b, nullptr, h_l);
    gemm(h_l, HHID, f_wih, biasc, gates, NNODES, 0);
    zero(st_l, 128);
    k_lstm<<<1, 256, 0, stream>>>(gates, whh, st_l, rnn_l);
    run_gin(h_l, HHID, HHID, ei1, f_g1w1, g1b1, f_g1w2, g1b2, bn1g, bn1b, h_l, h_l);
    gemm(h_l, HHID, f_wih, biasc, gates, NNODES, 0);
    k_lstm<<<1, 256, 0, stream>>>(gates, whh, st_l, rnn_l);

    // ---- side R ----
    run_gin(hin_r, CIN, CINP, ei2, f_g0w1, g0b1, f_g0w2, g0b2, bn0g, bn0b, nullptr, h_r);
    gemm(h_r, HHID, f_wih, biasc, gates, NNODES, 0);
    zero(st_r, 128);
    k_lstm<<<1, 256, 0, stream>>>(gates, whh, st_r, rnn_r);
    run_gin(h_r, HHID, HHID, ei2, f_g1w1, g1b1, f_g1w2, g1b2, bn1g, bn1b, h_r, h_r);
    gemm(h_r, HHID, f_wih, biasc, gates, NNODES, 0);
    k_lstm<<<1, 256, 0, stream>>>(gates, whh, st_r, rnn_r);

    // ---- attention pools: p2 on h (cols 0/64), p1 on rnn (cols 128/192) ----
    auto pool = [&](const float* sl, const float* sr,
                    const WF& wq, const float* bq, const WF& wk, const float* bk,
                    const WF& wv, const float* bv,
                    const float* lg, const float* lb, int offL, int offR) {
        gemm(sl, HHID, wq, bq, q1, NNODES, 1);
        gemm(sl, HHID, wk, bk, k1, NNODES, 1);
        gemm(sl, HHID, wv, bv, v1, NNODES, 1);
        gemm(sr, HHID, wq, bq, q2, NNODES, 1);
        gemm(sr, HHID, wk, bk, k2, NNODES, 1);
        gemm(sr, HHID, wv, bv, v2, NNODES, 1);
        k_attn<<<NB, 128, 0, stream>>>(k1, q2, v1, sl, m1, lg, lb, catb, offL);
        k_attn<<<NB, 128, 0, stream>>>(k2, q1, v2, sr, m2, lg, lb, catb, offR);
    };
    pool(h_l, h_r, f_p2q, p2bq, f_p2k, p2bk, f_p2v, p2bv, p2g, p2b, 0, 64);
    pool(rnn_l, rnn_r, f_p1q, p1bq, f_p1k, p1bk, f_p1v, p1bv, p1g, p1b, 128, 192);

    // ---- final MLP ----
    k_copy256<<<EL((long long)NB * 256), 256, 0, stream>>>(c256, catb);
    gemm(catb, 512, f_fw1, fb1, finmid, NB, 1);
    k_fin2<<<EL(NB * 2), 256, 0, stream>>>(finmid, fw2, fb2, out);
}
